// PointStructuringNet_31576599560764
// MI455X (gfx1250) — compile-verified, matrix operand order checked by reading, not run on hardware
//
#include <hip/hip_runtime.h>
#include <hip/hip_bf16.h>

typedef __attribute__((ext_vector_type(16))) _Float16 v16h;
typedef __attribute__((ext_vector_type(8)))  float    v8f;
typedef __attribute__((ext_vector_type(4)))  int      v4i_;

#define B_   4
#define N_   16384
#define C_   64
#define NP_  2048
#define KD_  256
#define NT_  1024   // N/16 column tiles
#define PT_  128    // NP/16 row tiles per batch
#define NQ_  4      // N-split factor (partial top-k quarters)
#define QT_  (NT_ / NQ_)

#define AS1 __attribute__((address_space(1)))
#define AS3 __attribute__((address_space(3)))

// ---------------------------------------------------------------------------
// CDNA5 async global->LDS copy (ASYNCcnt-tracked DMA), with portable fallback.
// ---------------------------------------------------------------------------
__device__ __forceinline__ void async_copy16(const _Float16* g, _Float16* l) {
#if __has_builtin(__builtin_amdgcn_global_load_async_to_lds_b128)
    __builtin_amdgcn_global_load_async_to_lds_b128(
        (AS1 v4i_*)(AS1 void*)g, (AS3 v4i_*)(AS3 void*)l, 0, 0);
#else
    *(int4*)l = *(const int4*)g;
#endif
}

__device__ __forceinline__ void wait_async() {
#if __has_builtin(__builtin_amdgcn_s_wait_asynccnt)
    __builtin_amdgcn_s_wait_asynccnt(0);
#elif defined(__AMDGCN__)
    asm volatile("s_wait_asynccnt 0x0" ::: "memory");
#endif
}

// ---------------------------------------------------------------------------
// Fold inference batchnorm into conv weights: w' = w*inv, b' = b - m*inv
// ---------------------------------------------------------------------------
__global__ void fold_bn(const float* __restrict__ w, const float* __restrict__ g,
                        const float* __restrict__ bb, const float* __restrict__ m,
                        const float* __restrict__ v, float* __restrict__ wf_,
                        float* __restrict__ bf_, int OC, int IC) {
    int c = blockIdx.x * blockDim.x + threadIdx.x;
    if (c >= OC) return;
    float inv = g[c] * rsqrtf(v[c] + 1e-5f);
    for (int k = 0; k < IC; ++k) wf_[c * IC + k] = w[c * IC + k] * inv;
    bf_[c] = bb[c] - m[c] * inv;
}

// ---------------------------------------------------------------------------
// Pack wf [2048,256] f32 -> f16 WMMA A-fragments.
//   lane<16 : M=lane,    K = i<8 ? i   : i+8
//   lane>=16: M=lane-16, K = i<8 ? i+8 : i+16
// ---------------------------------------------------------------------------
__global__ void pack_wf(const float* __restrict__ wf, _Float16* __restrict__ af) {
    int e    = blockIdx.x * 256 + threadIdx.x;   // 524288 elements
    int i    = e & 15;
    int lane = (e >> 4) & 31;
    int kt   = (e >> 9) & 7;
    int pt   = e >> 12;
    int M, K;
    if (lane < 16) { M = lane;      K = (i < 8) ? i     : i + 8;  }
    else           { M = lane - 16; K = (i < 8) ? i + 8 : i + 16; }
    af[e] = (_Float16)wf[(pt * 16 + M) * KD_ + kt * 32 + K];
}

// ---------------------------------------------------------------------------
// Fused point-MLP 3->64->128->256 (BN folded, ReLU) writing x directly as
// f16 WMMA B-fragments: xf[(((b*NT+nt)*8+kt)*32 + lane)*16 + i]
// ---------------------------------------------------------------------------
__global__ __launch_bounds__(64) void mlp_pack(
    const float* __restrict__ xyz,
    const float* __restrict__ w0f, const float* __restrict__ b0f,
    const float* __restrict__ w1f, const float* __restrict__ b1f,
    const float* __restrict__ w2f, const float* __restrict__ b2f,
    _Float16* __restrict__ xf) {
    __shared__ float h1[128 * 64];
    int tid = threadIdx.x;
    int pnt = blockIdx.x * 64 + tid;
    int b = pnt >> 14;
    int n = pnt & (N_ - 1);

    float x0 = xyz[pnt * 3 + 0], x1 = xyz[pnt * 3 + 1], x2 = xyz[pnt * 3 + 2];

    float h0[64];
#pragma unroll
    for (int c = 0; c < 64; ++c) {
        float a = fmaf(w0f[c * 3 + 0], x0,
                  fmaf(w0f[c * 3 + 1], x1,
                  fmaf(w0f[c * 3 + 2], x2, b0f[c])));
        h0[c] = fmaxf(a, 0.f);
    }
    for (int c = 0; c < 128; ++c) {
        float a = b1f[c];
#pragma unroll
        for (int k = 0; k < 64; ++k) a = fmaf(w1f[c * 64 + k], h0[k], a);
        h1[c * 64 + tid] = fmaxf(a, 0.f);
    }
    int nt = n >> 4, Nc = n & 15;
    size_t base = (size_t)(b * NT_ + nt) * 8 * 32 * 16;
    for (int c = 0; c < 256; ++c) {
        float a = b2f[c];
#pragma unroll 8
        for (int k = 0; k < 128; ++k) a = fmaf(w2f[c * 128 + k], h1[k * 64 + tid], a);
        a = fmaxf(a, 0.f);
        int kt = c >> 5, kk = c & 31;
        int lane, i;
        if      (kk < 8)  { lane = Nc;      i = kk;      }
        else if (kk < 16) { lane = Nc + 16; i = kk - 8;  }
        else if (kk < 24) { lane = Nc;      i = kk - 8;  }
        else              { lane = Nc + 16; i = kk - 16; }
        xf[base + (size_t)(kt * 32 + lane) * 16 + i] = (_Float16)a;
    }
}

// ---------------------------------------------------------------------------
// Fused WMMA GEMM + streaming partial top-32.
// Grid = 512 blocks: blockIdx = {q: N-quarter, g: M-tile group}. 4 waves/block,
// each wave owns a 16-row M-tile; 8KB B-tile staged once per block into LDS
// via double-buffered async DMA. Two independent WMMA accumulator chains.
// Emits unsorted partial 32-lists per (row, quarter) for the merge kernel.
// ---------------------------------------------------------------------------
__global__ __launch_bounds__(128) void gemm_topk(
    const _Float16* __restrict__ af, const _Float16* __restrict__ xf,
    float* __restrict__ pval, int* __restrict__ pidx) {
    __shared__ __align__(32) _Float16 bbuf[2][4096];   // 2 x 8KB B-tile stages
    __shared__ float s_val[4][16][32];
    __shared__ int   s_idx[4][16][32];
    __shared__ float s_thr[4][16];
    __shared__ int   s_cnt[4][16];

    int tid = threadIdx.x, w = tid >> 5, lane = tid & 31;
    int q  = blockIdx.x & 3;          // N-quarter
    int g  = blockIdx.x >> 2;         // 0..127 M-tile group
    int tw = g * 4 + w;               // 512 M-tiles total
    int b  = tw >> 7;
    int pt = tw & 127;

    if (lane < 16) { s_thr[w][lane] = -3.4e38f; s_cnt[w][lane] = 0; }

    v16h a[8];
#pragma unroll
    for (int kt = 0; kt < 8; ++kt)
        a[kt] = *(const v16h*)(af + ((size_t)(pt * 8 + kt) * 32 + lane) * 16);

    const _Float16* xb = xf + (size_t)b * NT_ * 4096;   // batch base
    int nt0 = q * QT_, nt1 = nt0 + QT_;

    // Stage first tile of this quarter.
#pragma unroll
    for (int j = 0; j < 4; ++j) {
        int e = tid + 128 * j;                           // 512 x 16B = 8KB
        async_copy16(xb + (size_t)nt0 * 4096 + (size_t)e * 8, &bbuf[0][e * 8]);
    }
    wait_async();
    __syncthreads();

    int rowbase = (lane >> 4) << 3;   // lanes 0-15 -> rows 0..7, 16-31 -> 8..15

    for (int nt = nt0; nt < nt1; ++nt) {
        int cur = nt & 1;
        if (nt + 1 < nt1) {           // kick DMA for next tile (overlaps WMMA)
            const _Float16* gs = xb + (size_t)(nt + 1) * 4096;
#pragma unroll
            for (int j = 0; j < 4; ++j) {
                int e = tid + 128 * j;
                async_copy16(gs + (size_t)e * 8, &bbuf[cur ^ 1][e * 8]);
            }
        }

        v8f c0 = {0.f, 0.f, 0.f, 0.f, 0.f, 0.f, 0.f, 0.f};
        v8f c1 = {0.f, 0.f, 0.f, 0.f, 0.f, 0.f, 0.f, 0.f};
#pragma unroll
        for (int kt = 0; kt < 8; kt += 2) {
            v16h b0v = *(const v16h*)&bbuf[cur][((kt + 0) * 32 + lane) * 16];
            v16h b1v = *(const v16h*)&bbuf[cur][((kt + 1) * 32 + lane) * 16];
            c0 = __builtin_amdgcn_wmma_f32_16x16x32_f16(
                     false, a[kt + 0], false, b0v, (short)0, c0, false, false);
            c1 = __builtin_amdgcn_wmma_f32_16x16x32_f16(
                     false, a[kt + 1], false, b1v, (short)0, c1, false, false);
        }
        v8f c = c0 + c1;

#pragma unroll
        for (int r = 0; r < 8; ++r) {
            float v = c[r];
            int rr = r + rowbase;
            float t = s_thr[w][rr];
            bool pass = v > t;
            unsigned long long mask = __ballot(pass);
            while (mask) {
                int s = __builtin_ctzll(mask);
                mask &= mask - 1;
                if (lane == s) {
                    float* vals = &s_val[w][rr][0];
                    int*   ids  = &s_idx[w][rr][0];
                    int ct = s_cnt[w][rr];
                    int col = nt * 16 + (lane & 15);
                    if (ct < 32) {
                        vals[ct] = v; ids[ct] = col; s_cnt[w][rr] = ct + 1;
                        if (ct == 31) {
                            float mn = vals[0];
                            for (int j = 1; j < 32; ++j) mn = fminf(mn, vals[j]);
                            s_thr[w][rr] = mn;
                        }
                    } else {
                        float mn = s_thr[w][rr];
                        int rp = 0;
                        for (int j = 0; j < 32; ++j) { if (vals[j] == mn) { rp = j; break; } }
                        vals[rp] = v; ids[rp] = col;
                        float nm = vals[0];
                        for (int j = 1; j < 32; ++j) nm = fminf(nm, vals[j]);
                        s_thr[w][rr] = nm;
                    }
                }
            }
        }

        wait_async();      // next tile's DMA has landed
        __syncthreads();   // all waves done reading bbuf[cur] before reuse
    }

    // Emit unsorted partial lists (always full: 4096 candidates per quarter).
    if (lane < 16) {
        int r = (b * 128 + pt) * 16 + lane;    // global row, 0..8191
        for (int j = 0; j < 32; ++j) {
            pval[(size_t)r * 128 + q * 32 + j] = s_val[w][lane][j];
            pidx[(size_t)r * 128 + q * 32 + j] = s_idx[w][lane][j];
        }
    }
}

// ---------------------------------------------------------------------------
// Merge 4 partial 32-lists per row -> final sorted top-32 (descending, ties:
// lower index). One wave per row; 4 candidates per lane; 32 rounds of
// wave-shuffle argmax reduction.
// ---------------------------------------------------------------------------
__global__ __launch_bounds__(256) void merge_topk(
    const float* __restrict__ pval, const int* __restrict__ pidx,
    int* __restrict__ idxout) {
    int lane = threadIdx.x & 31, wv = threadIdx.x >> 5;
    int row = blockIdx.x * 8 + wv;          // 8192 rows
    float v[4]; int id[4];
#pragma unroll
    for (int j = 0; j < 4; ++j) {
        v[j]  = pval[(size_t)row * 128 + 32 * j + lane];
        id[j] = pidx[(size_t)row * 128 + 32 * j + lane];
    }
    int res = 0;
    for (int s = 0; s < 32; ++s) {
        float bv = v[0]; int bi = id[0];
#pragma unroll
        for (int j = 1; j < 4; ++j)
            if (v[j] > bv || (v[j] == bv && id[j] < bi)) { bv = v[j]; bi = id[j]; }
        float rv = bv; int ri = bi;
#pragma unroll
        for (int off = 16; off >= 1; off >>= 1) {
            float ov = __shfl_xor(rv, off, 32);
            int   oi = __shfl_xor(ri, off, 32);
            if (ov > rv || (ov == rv && oi < ri)) { rv = ov; ri = oi; }
        }
        // invalidate the (unique) winner
#pragma unroll
        for (int j = 0; j < 4; ++j)
            if (v[j] == rv && id[j] == ri) { v[j] = -3.4e38f; id[j] = 0x7fffffff; }
        if (s == lane) res = ri;
    }
    idxout[(size_t)row * 32 + lane] = res;
}

// ---------------------------------------------------------------------------
// Gather epilogue: out[b, 0:3, p, s] = xyz_trans, out[b, 3:67, p, s] = features
// ---------------------------------------------------------------------------
__global__ __launch_bounds__(256) void gather_out(
    const float* __restrict__ xyz, const float* __restrict__ feat,
    const int* __restrict__ idxbuf, float* __restrict__ out) {
    int tid = threadIdx.x, s = tid & 31, g = tid >> 5;
    int pair = blockIdx.x * 8 + g;     // 8192 = B*NP
    int b = pair >> 11, p = pair & 2047;
    int n = idxbuf[((size_t)b * NP_ + p) * 32 + s];
    size_t ob = (size_t)b * 67 * NP_ * 32 + (size_t)p * 32 + s;
    for (int c = 0; c < 67; ++c) {
        float v;
        if (c < 3) v = xyz[((size_t)b * N_ + n) * 3 + c];
        else       v = feat[((size_t)(b * C_) + (c - 3)) * (size_t)N_ + n];
        out[ob + (size_t)c * NP_ * 32] = v;
    }
}

// ---------------------------------------------------------------------------
extern "C" void kernel_launch(void* const* d_in, const int* in_sizes, int n_in,
                              void* d_out, int out_size, void* d_ws, size_t ws_size,
                              hipStream_t stream) {
    const float* xyz  = (const float*)d_in[0];
    const float* feat = (const float*)d_in[1];
    const float* w0 = (const float*)d_in[2];
    const float* g0 = (const float*)d_in[3];
    const float* b0 = (const float*)d_in[4];
    const float* m0 = (const float*)d_in[5];
    const float* v0 = (const float*)d_in[6];
    const float* w1 = (const float*)d_in[7];
    const float* g1 = (const float*)d_in[8];
    const float* b1 = (const float*)d_in[9];
    const float* m1 = (const float*)d_in[10];
    const float* v1 = (const float*)d_in[11];
    const float* w2 = (const float*)d_in[12];
    const float* g2 = (const float*)d_in[13];
    const float* b2 = (const float*)d_in[14];
    const float* m2 = (const float*)d_in[15];
    const float* v2 = (const float*)d_in[16];
    const float* wf = (const float*)d_in[17];
    // d_in[18] = bf is provably irrelevant: per-row constant + monotone sigmoid
    // cannot change top-k indices.
    float* out = (float*)d_out;

    char* wsb = (char*)d_ws;
    _Float16* xf   = (_Float16*)(wsb + 0);               // 32 MB  B-fragments
    _Float16* af   = (_Float16*)(wsb + 33554432);        // 1 MB   A-fragments
    float*    pval = (float*)   (wsb + 34603008);        // 4 MB   partial vals
    int*      pidx = (int*)     (wsb + 38797312);        // 4 MB   partial idx
    int*      idxb = (int*)     (wsb + 42991616);        // 1 MB   final indices
    float*    w0f  = (float*)   (wsb + 44040192);
    float*    b0f  = (float*)   (wsb + 44041216);
    float*    w1f  = (float*)   (wsb + 44041472);
    float*    b1f  = (float*)   (wsb + 44074240);
    float*    w2f  = (float*)   (wsb + 44074752);
    float*    b2f  = (float*)   (wsb + 44205824);

    fold_bn<<<1, 64,  0, stream>>>(w0, g0, b0, m0, v0, w0f, b0f, 64, 3);
    fold_bn<<<1, 128, 0, stream>>>(w1, g1, b1, m1, v1, w1f, b1f, 128, 64);
    fold_bn<<<1, 256, 0, stream>>>(w2, g2, b2, m2, v2, w2f, b2f, 256, 128);

    pack_wf<<<2048, 256, 0, stream>>>(wf, af);

    mlp_pack<<<(B_ * N_) / 64, 64, 0, stream>>>(xyz, w0f, b0f, w1f, b1f, w2f, b2f, xf);

    gemm_topk<<<(B_ * PT_ / 4) * NQ_, 128, 0, stream>>>(af, xf, pval, pidx);

    merge_topk<<<(B_ * NP_) / 8, 256, 0, stream>>>(pval, pidx, idxb);

    gather_out<<<(B_ * NP_) / 8, 256, 0, stream>>>(xyz, feat, idxb, out);
}
